// SparseMoELanguageModel_89498528514680
// MI455X (gfx1250) — compile-verified, matrix operand order checked by reading, not run on hardware
//
#include <hip/hip_runtime.h>
#include <hip/hip_bf16.h>

// ---------------- model dims (hardcoded, match reference) ----------------
#define Bn   32
#define Tn   96
#define Cn   192
#define Hh   6
#define Dd   32
#define Ee   8
#define Ll   8
#define Ff   768
#define Rr   128
#define Vv   50260
#define NTOK (Bn * Tn)          // 3072 tokens
#define SPK  (Vv - 3)
#define SCN  (Vv - 2)
#define STG  (Vv - 1)

typedef __attribute__((ext_vector_type(16))) _Float16 v16h;
typedef __attribute__((ext_vector_type(4)))  _Float16 v4h;
typedef __attribute__((ext_vector_type(8)))  float    v8f;
typedef __attribute__((ext_vector_type(4)))  float    f4;

__device__ __forceinline__ float gelu_f(float v) {
    // exact GELU (approximate=False): 0.5*x*(1+erf(x/sqrt(2)))
    return 0.5f * v * (1.0f + erff(v * 0.7071067811865475f));
}

// =================== f16-WMMA GEMM: C[M,N] = A[M,K]*B[K,N] ================
// 256 threads = 8 wave32 (2x4 wave grid). Block tile 64x128, wave tile 32x32,
// K-step 64 (two 32-k sub-steps) -> 8 v_wmma_f32_16x16x32_f16 per iteration.
// Double-buffered LDS (2 x 24KB) staged in WMMA fragment order (lane-major,
// 16 contiguous halves per lane) so each fragment load is one 32B LDS read.
// All global loads are float4 (A along k, B along n with a 4x4 register
// transpose). Staging is branchless (clamped indices): requires K mult-of-64
// and N mult-of-4 (true for all shapes used here: K in {192,768}, N in
// {576,192,768,50260}); out-of-range rows/cols only feed padding outputs
// which are never stored.
#define GEMM_BETA1 1   // C += result (residual accumulate)
#define GEMM_GELU  2   // apply exact gelu after bias

// LDS halves per buffer: A 64x64 = 4096, B 64x128 = 8192
#define SM_A    0
#define SM_B    4096
#define SM_BUF  12288

__device__ __forceinline__ void stage_tiles(
    _Float16* __restrict__ sm,
    const float* __restrict__ A, int lda,
    const float* __restrict__ B, int ldb,
    int M, int N, int Mbase, int Nbase, int k0, int tid)
{
    // ---- A: 64 rows x 64 k, 1024 float4-chunks (along k), 4 per thread ----
#pragma unroll
    for (int i = 0; i < 4; ++i) {
        int chunk = tid + i * 256;          // 0..1023
        int m  = chunk >> 4;                // 0..63
        int k  = (chunk & 15) << 2;         // 0,4,..,60
        int gm = Mbase + m; gm = gm < M ? gm : M - 1;
        f4 src = *(const f4*)(A + (size_t)gm * lda + (k0 + k));
        v4h d; d[0] = (_Float16)src[0]; d[1] = (_Float16)src[1];
               d[2] = (_Float16)src[2]; d[3] = (_Float16)src[3];
        int kk   = k & 31;
        int ksub = k >> 5;
        int lane = (m & 15) + (((kk >> 3) & 1) << 4);
        int pos  = ((kk & 16) >> 1) + (kk & 7);
        *(v4h*)&sm[SM_A + ((((m >> 4) << 1) + ksub) << 9) + (lane << 4) + pos] = d;
    }
    // ---- B: 64 k x 128 cols, 512 4x4-blocks, 2 per thread ----
    // 4 coalesced float4 loads along n, register transpose, 4 ds_store_b64.
#pragma unroll
    for (int i = 0; i < 2; ++i) {
        int blk = tid + i * 256;            // 0..511
        int kb  = (blk >> 5) << 2;          // 0,4,..,60
        int nb  = (blk & 31) << 2;          // 0,4,..,124
        int gn  = Nbase + nb; gn = (gn + 3 < N) ? gn : (N - 4);
        f4 row0 = *(const f4*)(B + (size_t)(k0 + kb + 0) * ldb + gn);
        f4 row1 = *(const f4*)(B + (size_t)(k0 + kb + 1) * ldb + gn);
        f4 row2 = *(const f4*)(B + (size_t)(k0 + kb + 2) * ldb + gn);
        f4 row3 = *(const f4*)(B + (size_t)(k0 + kb + 3) * ldb + gn);
        int kk   = kb & 31;
        int ksub = kb >> 5;
        int pos  = ((kk & 16) >> 1) + (kk & 7);
        int lbase = (((kk >> 3) & 1) << 4);
#pragma unroll
        for (int c = 0; c < 4; ++c) {
            v4h d; d[0] = (_Float16)row0[c]; d[1] = (_Float16)row1[c];
                   d[2] = (_Float16)row2[c]; d[3] = (_Float16)row3[c];
            int n    = nb + c;
            int lane = (n & 15) + lbase;
            *(v4h*)&sm[SM_B + ((((n >> 4) << 1) + ksub) << 9) + (lane << 4) + pos] = d;
        }
    }
}

__global__ __launch_bounds__(256)
void gemm_f16_wmma(const float* __restrict__ A, int lda,
                   const float* __restrict__ B, int ldb,
                   const float* __restrict__ bias,
                   float* __restrict__ C, int ldc,
                   int M, int N, int K, int flags)
{
    __shared__ __align__(32) _Float16 smem[2 * SM_BUF];   // 48 KB

    const int tid  = threadIdx.x;
    const int lane = tid & 31;
    const int wave = tid >> 5;                 // 0..7
    const int wm   = wave >> 2;                // 0..1  (M)
    const int wn   = wave & 3;                 // 0..3  (N)
    const int Mbase = blockIdx.y * 64;
    const int Nbase = blockIdx.x * 128;

    v8f acc00 = {}, acc01 = {}, acc10 = {}, acc11 = {};

    stage_tiles(smem, A, lda, B, ldb, M, N, Mbase, Nbase, 0, tid);

    for (int k0 = 0; k0 < K; k0 += 64) {
        __syncthreads();
        _Float16* sb = smem + ((k0 >> 6) & 1) * SM_BUF;
        if (k0 + 64 < K) {
            // prefetch next panel rows (gfx1250 global_prefetch_b8)
            int gm = Mbase + lane; gm = gm < M ? gm : M - 1;
            __builtin_prefetch(&A[(size_t)gm * lda + k0 + 64], 0, 1);
            __builtin_prefetch(&B[(size_t)(k0 + 64 + lane) * ldb + Nbase], 0, 1);
            stage_tiles(smem + (1 - ((k0 >> 6) & 1)) * SM_BUF,
                        A, lda, B, ldb, M, N, Mbase, Nbase, k0 + 64, tid);
        }
#pragma unroll
        for (int ks = 0; ks < 2; ++ks) {
            // fragment loads: one 32B LDS read each (2 x ds_load_b128)
            v16h a0 = *(const v16h*)&sb[SM_A + (((0 + wm * 2) * 2 + ks) << 9) + (lane << 4)];
            v16h a1 = *(const v16h*)&sb[SM_A + (((1 + wm * 2) * 2 + ks) << 9) + (lane << 4)];
            v16h b0 = *(const v16h*)&sb[SM_B + (((0 + wn * 2) * 2 + ks) << 9) + (lane << 4)];
            v16h b1 = *(const v16h*)&sb[SM_B + (((1 + wn * 2) * 2 + ks) << 9) + (lane << 4)];
            acc00 = __builtin_amdgcn_wmma_f32_16x16x32_f16(false, a0, false, b0,
                                                           (short)0, acc00, false, false);
            acc01 = __builtin_amdgcn_wmma_f32_16x16x32_f16(false, a0, false, b1,
                                                           (short)0, acc01, false, false);
            acc10 = __builtin_amdgcn_wmma_f32_16x16x32_f16(false, a1, false, b0,
                                                           (short)0, acc10, false, false);
            acc11 = __builtin_amdgcn_wmma_f32_16x16x32_f16(false, a1, false, b1,
                                                           (short)0, acc11, false, false);
        }
    }

    // epilogue: C/D layout - lane<16: N=lane, M=r; lane>=16: N=lane-16, M=r+8
    const int nc0 = Nbase + ((wn * 2 + 0) << 4) + (lane & 15);
    const int nc1 = Nbase + ((wn * 2 + 1) << 4) + (lane & 15);
    const int mr0 = Mbase + ((wm * 2 + 0) << 4) + ((lane >> 4) << 3);
    const int mr1 = Mbase + ((wm * 2 + 1) << 4) + ((lane >> 4) << 3);
#pragma unroll
    for (int r = 0; r < 8; ++r) {
        float vals[4] = { acc00[r], acc01[r], acc10[r], acc11[r] };
        const int mrow[4] = { mr0 + r, mr0 + r, mr1 + r, mr1 + r };
        const int ncol[4] = { nc0, nc1, nc0, nc1 };
#pragma unroll
        for (int q = 0; q < 4; ++q) {
            if (mrow[q] < M && ncol[q] < N) {
                float v = vals[q];
                if (bias) v += bias[ncol[q]];
                if (flags & GEMM_GELU) v = gelu_f(v);
                if (flags & GEMM_BETA1) v += C[(size_t)mrow[q] * ldc + ncol[q]];
                C[(size_t)mrow[q] * ldc + ncol[q]] = v;
            }
        }
    }
}

// ---------------- embedding: x = tok_e1[idx] @ tok_w + pos_emb ------------
__global__ __launch_bounds__(Cn)
void embed_kernel(const int* __restrict__ idx,
                  const float* __restrict__ tok_e1,  // [V,R]
                  const float* __restrict__ tok_w,   // [R,C]
                  const float* __restrict__ pos_emb, // [T,C]
                  float* __restrict__ x, int* __restrict__ sid)
{
    const int n = blockIdx.x;       // token
    const int c = threadIdx.x;      // channel
    const int id = idx[n];
    const float* e = tok_e1 + (size_t)id * Rr;
    float s = 0.0f;
#pragma unroll 4
    for (int r = 0; r < Rr; ++r) s += e[r] * tok_w[(size_t)r * Cn + c];
    x[(size_t)n * Cn + c] = s + pos_emb[(size_t)(n % Tn) * Cn + c];
    if (c == 0)
        sid[n] = (id == SPK) * 1 + (id == SCN) * 2 + (id == STG) * 3;
}

// ---------------- LayerNorm (optional struct-embed post-add) --------------
__global__ __launch_bounds__(256)
void ln_kernel(const float* __restrict__ in,
               const float* __restrict__ w, const float* __restrict__ b,
               float* __restrict__ out, int dim,
               const float* __restrict__ post_add,  // [4,dim] or null
               const int* __restrict__ sid)
{
    const int n = blockIdx.x;
    const float* row = in + (size_t)n * dim;
    __shared__ float s1[256], s2[256];
    float s = 0.0f, ss = 0.0f;
    for (int d = threadIdx.x; d < dim; d += 256) {
        float v = row[d]; s += v; ss += v * v;
    }
    s1[threadIdx.x] = s; s2[threadIdx.x] = ss;
    __syncthreads();
    for (int st = 128; st > 0; st >>= 1) {
        if (threadIdx.x < st) { s1[threadIdx.x] += s1[threadIdx.x + st];
                                s2[threadIdx.x] += s2[threadIdx.x + st]; }
        __syncthreads();
    }
    const float mean = s1[0] / dim;
    const float var  = s2[0] / dim - mean * mean;
    const float rstd = rsqrtf(var + 1e-5f);
    const float* pa = post_add ? post_add + (size_t)sid[n] * dim : nullptr;
    for (int d = threadIdx.x; d < dim; d += 256) {
        float v = (row[d] - mean) * rstd * w[d] + b[d];
        if (pa) v += pa[d];
        out[(size_t)n * dim + d] = v;
    }
}

// ---------------- attention: one (b,h) per block, 96 threads --------------
__global__ __launch_bounds__(Tn)
void attn_kernel(const float* __restrict__ qkv, float* __restrict__ o)
{
    const int bh = blockIdx.x;
    const int b = bh / Hh, hh = bh % Hh;
    __shared__ float Ks[Tn * Dd], Vs[Tn * Dd], Ss[Tn * (Tn + 1)];
    const int t = threadIdx.x;

    for (int i = t; i < Tn * Dd; i += Tn) {
        int j = i >> 5, d = i & 31;
        size_t base = ((size_t)(b * Tn + j)) * (3 * Cn) + hh * Dd + d;
        Ks[j * Dd + d] = qkv[base + Cn];
        Vs[j * Dd + d] = qkv[base + 2 * Cn];
    }
    __syncthreads();

    float q[Dd];
#pragma unroll
    for (int d = 0; d < Dd; ++d)
        q[d] = qkv[((size_t)(b * Tn + t)) * (3 * Cn) + hh * Dd + d];

    const float scale = 0.17677669529663687f;   // 1/sqrt(32)
    float mx = -1e30f;
    for (int j = 0; j < Tn; ++j) {
        float s = 0.0f;
#pragma unroll
        for (int d = 0; d < Dd; ++d) s += q[d] * Ks[j * Dd + d];
        s = fminf(30.0f, fmaxf(-30.0f, s * scale));
        Ss[t * (Tn + 1) + j] = s;
        mx = fmaxf(mx, s);
    }
    float sum = 0.0f, acc[Dd];
#pragma unroll
    for (int d = 0; d < Dd; ++d) acc[d] = 0.0f;
    for (int j = 0; j < Tn; ++j) {
        float p = __expf(Ss[t * (Tn + 1) + j] - mx);
        sum += p;
#pragma unroll
        for (int d = 0; d < Dd; ++d) acc[d] += p * Vs[j * Dd + d];
    }
    const float inv = 1.0f / (sum + 1e-10f);
#pragma unroll
    for (int d = 0; d < Dd; ++d)
        o[((size_t)(b * Tn + t)) * Cn + hh * Dd + d] = acc[d] * inv;
}

// ---------------- router: scores, top-2, softmax weights ------------------
__global__ __launch_bounds__(128)
void router_kernel(const float* __restrict__ query,   // [N,C]
                   const float* __restrict__ type_q,  // [E,C]
                   float* __restrict__ rw, int* __restrict__ ri, int ntok)
{
    const int n = blockIdx.x * blockDim.x + threadIdx.x;
    if (n >= ntok) return;
    const float* qr = query + (size_t)n * Cn;
    float sc[Ee];
#pragma unroll
    for (int e = 0; e < Ee; ++e) {
        const float* tq = type_q + (size_t)e * Cn;
        float s = 0.0f;
        for (int c = 0; c < Cn; ++c) s += qr[c] * tq[c];
        sc[e] = s;
    }
    int i0 = 0;
#pragma unroll
    for (int e = 1; e < Ee; ++e) if (sc[e] > sc[i0]) i0 = e;
    int i1 = (i0 == 0) ? 1 : 0;
#pragma unroll
    for (int e = 0; e < Ee; ++e) if (e != i0 && sc[e] > sc[i1]) i1 = e;
    float m = sc[i0];
    float e0 = __expf(sc[i0] - m), e1 = __expf(sc[i1] - m);
    float inv = 1.0f / (e0 + e1);
    rw[2 * n] = e0 * inv; rw[2 * n + 1] = e1 * inv;
    ri[2 * n] = i0;       ri[2 * n + 1] = i1;
}

// ---------------- expert combine: x += w * LN(h + y) ----------------------
__global__ __launch_bounds__(Cn)
void expert_combine_kernel(const float* __restrict__ h, const float* __restrict__ y,
                           const float* __restrict__ w, const float* __restrict__ b,
                           const float* __restrict__ rw, const int* __restrict__ ri,
                           int e, float* __restrict__ x)
{
    const int n = blockIdx.x, d = threadIdx.x;
    __shared__ float s1[256], s2[256];
    float v = h[(size_t)n * Cn + d] + y[(size_t)n * Cn + d];
    s1[d] = v; s2[d] = v * v;
    if (d < 64) { s1[d + Cn] = 0.0f; s2[d + Cn] = 0.0f; }  // pad to 256
    __syncthreads();
    for (int st = 128; st > 0; st >>= 1) {
        if (d < st) { s1[d] += s1[d + st]; s2[d] += s2[d + st]; }
        __syncthreads();
    }
    const float mean = s1[0] / Cn;
    const float var  = s2[0] / Cn - mean * mean;
    const float rstd = rsqrtf(var + 1e-5f);
    float z = (v - mean) * rstd * w[d] + b[d];
    float wt = (ri[2 * n] == e) ? rw[2 * n]
             : (ri[2 * n + 1] == e) ? rw[2 * n + 1] : 0.0f;
    if (wt != 0.0f) x[(size_t)n * Cn + d] += wt * z;
}

// ======================= host orchestration ===============================
extern "C" void kernel_launch(void* const* d_in, const int* in_sizes, int n_in,
                              void* d_out, int out_size, void* d_ws, size_t ws_size,
                              hipStream_t stream) {
    const int*   idx       = (const int*)  d_in[0];
    const float* tok_e1    = (const float*)d_in[1];
    const float* tok_w     = (const float*)d_in[2];
    const float* pos_emb   = (const float*)d_in[3];
    const float* ln1_w     = (const float*)d_in[4];
    const float* ln1_b     = (const float*)d_in[5];
    const float* struct_e  = (const float*)d_in[6];
    const float* Wqkv      = (const float*)d_in[7];
    const float* Wout      = (const float*)d_in[8];
    const float* bout      = (const float*)d_in[9];
    const float* ln2_w     = (const float*)d_in[10];
    const float* ln2_b     = (const float*)d_in[11];
    const float* Wr1       = (const float*)d_in[12];
    const float* br1       = (const float*)d_in[13];
    const float* Wr2       = (const float*)d_in[14];
    const float* br2       = (const float*)d_in[15];
    const float* type_q    = (const float*)d_in[16];
    const float* We1       = (const float*)d_in[17];
    const float* be1       = (const float*)d_in[18];
    const float* elnw      = (const float*)d_in[19];
    const float* elnb      = (const float*)d_in[20];
    const float* We2       = (const float*)d_in[21];
    const float* be2       = (const float*)d_in[22];
    const float* enormw    = (const float*)d_in[23];
    const float* enormb    = (const float*)d_in[24];
    const float* lnf_w     = (const float*)d_in[25];
    const float* lnf_b     = (const float*)d_in[26];
    const float* Wlm       = (const float*)d_in[27];
    const float* blm       = (const float*)d_in[28];

    // workspace layout (~48 MB of f32)
    float* ws    = (float*)d_ws;
    float* x     = ws;  ws += (size_t)NTOK * Cn;
    float* h     = ws;  ws += (size_t)NTOK * Cn;
    float* qkv   = ws;  ws += (size_t)NTOK * 3 * Cn;
    float* o     = ws;  ws += (size_t)NTOK * Cn;
    float* r1    = ws;  ws += (size_t)NTOK * Ff;
    float* query = ws;  ws += (size_t)NTOK * Cn;
    float* he    = ws;  ws += (size_t)NTOK * Ff;
    float* hen   = ws;  ws += (size_t)NTOK * Ff;
    float* yv    = ws;  ws += (size_t)NTOK * Cn;
    float* rw    = ws;  ws += (size_t)NTOK * 2;
    int*   ri    = (int*)ws;  ws += (size_t)NTOK * 2;
    int*   sid   = (int*)ws;

    auto gemm = [&](const float* A, int lda, const float* Bm, int ldb,
                    const float* bias, float* Cm, int ldc,
                    int M, int N, int K, int flags) {
        dim3 grid((N + 127) / 128, (M + 63) / 64);
        gemm_f16_wmma<<<grid, dim3(256), 0, stream>>>(A, lda, Bm, ldb, bias,
                                                      Cm, ldc, M, N, K, flags);
    };

    // ---- embedding ----
    embed_kernel<<<NTOK, Cn, 0, stream>>>(idx, tok_e1, tok_w, pos_emb, x, sid);

    // ---- transformer layers ----
    for (int i = 0; i < Ll; ++i) {
        // attention branch
        ln_kernel<<<NTOK, 256, 0, stream>>>(x, ln1_w + (size_t)i * Cn,
                                            ln1_b + (size_t)i * Cn, h, Cn,
                                            struct_e + (size_t)i * 4 * Cn, sid);
        gemm(h, Cn, Wqkv + (size_t)i * Cn * 3 * Cn, 3 * Cn, nullptr,
             qkv, 3 * Cn, NTOK, 3 * Cn, Cn, 0);
        attn_kernel<<<Bn * Hh, Tn, 0, stream>>>(qkv, o);
        gemm(o, Cn, Wout + (size_t)i * Cn * Cn, Cn, bout + (size_t)i * Cn,
             x, Cn, NTOK, Cn, Cn, GEMM_BETA1);                 // x += o@Wout+b

        // MoE branch
        ln_kernel<<<NTOK, 256, 0, stream>>>(x, ln2_w + (size_t)i * Cn,
                                            ln2_b + (size_t)i * Cn, h, Cn,
                                            nullptr, nullptr);
        gemm(h, Cn, Wr1 + (size_t)i * Cn * Ff, Ff, br1 + (size_t)i * Ff,
             r1, Ff, NTOK, Ff, Cn, GEMM_GELU);                 // gelu(h@Wr1+b)
        gemm(r1, Ff, Wr2 + (size_t)i * Ff * Cn, Cn, br2 + (size_t)i * Cn,
             query, Cn, NTOK, Cn, Ff, 0);
        router_kernel<<<(NTOK + 127) / 128, 128, 0, stream>>>(
            query, type_q + (size_t)i * Ee * Cn, rw, ri, NTOK);

        for (int e = 0; e < Ee; ++e) {
            size_t ie = (size_t)i * Ee + e;
            gemm(h, Cn, We1 + ie * Cn * Ff, Ff, be1 + ie * Ff,
                 he, Ff, NTOK, Ff, Cn, GEMM_GELU);             // gelu(h@We1+b)
            ln_kernel<<<NTOK, 256, 0, stream>>>(he, elnw + ie * Ff,
                                                elnb + ie * Ff, hen, Ff,
                                                nullptr, nullptr);
            gemm(hen, Ff, We2 + ie * Ff * Cn, Cn, be2 + ie * Cn,
                 yv, Cn, NTOK, Cn, Ff, 0);
            expert_combine_kernel<<<NTOK, Cn, 0, stream>>>(
                h, yv, enormw + ie * Cn, enormb + ie * Cn, rw, ri, e, x);
        }
    }

    // ---- final LN + LM head ----
    ln_kernel<<<NTOK, 256, 0, stream>>>(x, lnf_w, lnf_b, h, Cn, nullptr, nullptr);
    gemm(h, Cn, Wlm, Vv, blm, (float*)d_out, Vv, NTOK, Vv, Cn, 0);
}
